// GAT_dm_89481348645414
// MI455X (gfx1250) — compile-verified, matrix operand order checked by reading, not run on hardware
//
#include <hip/hip_runtime.h>
#include <hip/hip_bf16.h>

typedef _Float16 v16h __attribute__((ext_vector_type(16)));
typedef _Float16 v8h  __attribute__((ext_vector_type(8)));
typedef float    v8f  __attribute__((ext_vector_type(8)));

#define EPS 1e-5f

// ---------------- wave32 reductions ----------------
__device__ __forceinline__ float wredsum(float v) {
#pragma unroll
    for (int o = 16; o; o >>= 1) v += __shfl_xor(v, o, 32);
    return v;
}
__device__ __forceinline__ float wredmax(float v) {
#pragma unroll
    for (int o = 16; o; o >>= 1) v = fmaxf(v, __shfl_xor(v, o, 32));
    return v;
}

__device__ __forceinline__ void load18(const float* __restrict__ p, float* r) {
    float4 a0 = *(const float4*)(p);
    float4 a1 = *(const float4*)(p + 4);
    float4 a2 = *(const float4*)(p + 8);
    float4 a3 = *(const float4*)(p + 12);
    float2 a4 = *(const float2*)(p + 16);
    r[0]=a0.x; r[1]=a0.y; r[2]=a0.z; r[3]=a0.w;
    r[4]=a1.x; r[5]=a1.y; r[6]=a1.z; r[7]=a1.w;
    r[8]=a2.x; r[9]=a2.y; r[10]=a2.z; r[11]=a2.w;
    r[12]=a3.x; r[13]=a3.y; r[14]=a3.z; r[15]=a3.w;
    r[16]=a4.x; r[17]=a4.y;
}

// ---------------- column stats (sum, sumsq) ----------------
template<int C, int S>
__global__ void stats_kernel(const float* __restrict__ v, float* __restrict__ st, int n) {
    __shared__ float ls[2 * C];
    const int tid = threadIdx.x;
    if (tid < 2 * C) ls[tid] = 0.0f;
    __syncthreads();
    float s[C]; float q[C];
#pragma unroll
    for (int j = 0; j < C; ++j) { s[j] = 0.0f; q[j] = 0.0f; }
    for (int row = blockIdx.x * blockDim.x + tid; row < n; row += gridDim.x * blockDim.x) {
#pragma unroll
        for (int j = 0; j < C; ++j) {
            float x = v[(size_t)row * S + j];
            s[j] += x; q[j] += x * x;
        }
    }
#pragma unroll
    for (int j = 0; j < C; ++j) {
        atomicAdd(&ls[j], s[j]);
        atomicAdd(&ls[C + j], q[j]);
    }
    __syncthreads();
    if (tid < 2 * C) atomicAdd(&st[tid], ls[tid]);
}

// ---------------- BN0 apply (scalar column) ----------------
__global__ void bn0_kernel(const float* __restrict__ x, const float* __restrict__ st,
                           const float* __restrict__ g, const float* __restrict__ be,
                           float* __restrict__ x0bn, int n) {
    int i = blockIdx.x * blockDim.x + threadIdx.x;
    if (i >= n) return;
    float inv_n = 1.0f / (float)n;
    float mu = st[0] * inv_n;
    float var = st[1] * inv_n - mu * mu;
    x0bn[i] = (x[i] - mu) * rsqrtf(var + EPS) * g[0] + be[0];
}

// ---------------- layer 0 edge kernel (1-channel input => scalar algebra) ----------
// score_h = A_h*v_s + B_h*v_d + C_h + ef.aW[12:14];  y[h,f] = (W0[h,f]*m_h + b0[h,f])/K
__global__ void edge0_kernel(const float* __restrict__ x0bn,
                             const int* __restrict__ ei,
                             const float* __restrict__ ef,   // [N*32][2]
                             const float* __restrict__ aW,   // [14]
                             const float* __restrict__ ab,   // [1]
                             const float* __restrict__ W0,   // [18]
                             const float* __restrict__ b0,   // [18]
                             float* __restrict__ y,          // [N][20]
                             int n) {
    const int wave = threadIdx.x >> 5;
    const int lane = threadIdx.x & 31;
    const int node = blockIdx.x * (blockDim.x >> 5) + wave;
    if (node >= n) return;                 // wave-uniform
    const int e = node * 32 + lane;
    const int s = ei[e];
    const int d = ei[(size_t)n * 32 + e];

    const float vs = x0bn[s];
    const float vd = x0bn[d];
    const float2 ev = *(const float2*)(ef + (size_t)e * 2);
    const float efc = ev.x * aW[12] + ev.y * aW[13];
    const float abv = ab[0];

    float m[3];
#pragma unroll
    for (int h = 0; h < 3; ++h) {
        // per-head uniform constants (scalar loads; uniform across wave)
        float A = 0.0f, B = 0.0f, C = abv;
#pragma unroll
        for (int f = 0; f < 6; ++f) {
            A += W0[h * 6 + f] * aW[f];
            B += W0[h * 6 + f] * aW[6 + f];
            C += b0[h * 6 + f] * (aW[f] + aW[6 + f]);
        }
        float sc = A * vs + B * vd + C + efc;
        float mx = wredmax(sc);
        float p = __expf(sc - mx);
        float ssum = wredsum(p);
        m[h] = wredsum(p * vs) / ssum;     // sum_k att_{k,h} * v_{s_k}
    }

    if (lane == 0) {
        float* py = y + (size_t)node * 20;
#pragma unroll
        for (int h = 0; h < 3; ++h)
#pragma unroll
            for (int f = 0; f < 6; ++f)
                py[h * 6 + f] = (W0[h * 6 + f] * m[h] + b0[h * 6 + f]) * (1.0f / 32.0f);
    }
}

// ---------------- drain-side projection: pd[n][h] = sum_f xe[n][h,f]*aW[6+f] --------
__global__ void proj_pd_kernel(const float* __restrict__ xe, const float* __restrict__ aW,
                               float4* __restrict__ pd, int n) {
    int i = blockIdx.x * blockDim.x + threadIdx.x;
    if (i >= n) return;
    float r[18];
    load18(xe + (size_t)i * 20, r);
    float4 o = {0.0f, 0.0f, 0.0f, 0.0f};
#pragma unroll
    for (int f = 0; f < 6; ++f) {
        float w = aW[6 + f];
        o.x += r[f] * w;
        o.y += r[6 + f] * w;
        o.z += r[12 + f] * w;
    }
    pd[i] = o;
}

// ---------------- edge attention, layers 1/2: one wave32 per node -------------------
__global__ void edge_attn_kernel(const float* __restrict__ xe,   // [N][20]
                                 const float4* __restrict__ pd,  // [N] drain projections
                                 const int* __restrict__ ei,     // [2][N*32]
                                 const float* __restrict__ aW,   // [12]
                                 const float* __restrict__ ab,
                                 float* __restrict__ y,          // [N][20]
                                 int n) {
    const int wave = threadIdx.x >> 5;
    const int lane = threadIdx.x & 31;
    const int node = blockIdx.x * (blockDim.x >> 5) + wave;
    if (node >= n) return;                 // wave-uniform
    const int e = node * 32 + lane;
    const int s = ei[e];
    const int d = ei[(size_t)n * 32 + e];

    float xs[18];
    load18(xe + (size_t)s * 20, xs);
    const float4 pdv = pd[d];
    const float pdl[3] = {pdv.x, pdv.y, pdv.z};
    const float abv = ab[0];

    float att[3];
#pragma unroll
    for (int h = 0; h < 3; ++h) {
        float sc = abv + pdl[h];
#pragma unroll
        for (int f = 0; f < 6; ++f)
            sc += xs[h * 6 + f] * aW[f];   // src-side projection in-register
        float mx = wredmax(sc);
        float p = __expf(sc - mx);
        float ssum = wredsum(p);
        att[h] = p / ssum;
    }

    float out[18];
#pragma unroll
    for (int h = 0; h < 3; ++h)
#pragma unroll
        for (int f = 0; f < 6; ++f)
            out[h * 6 + f] = wredsum(att[h] * xs[h * 6 + f]) * (1.0f / 32.0f);

    if (lane == 0) {
        float* py = y + (size_t)node * 20;
#pragma unroll
        for (int j = 0; j < 18; ++j) py[j] = out[j];
    }
}

// ---------------- BN apply -> f16 padded input buffers ----------------
__global__ void bn1_buf1_kernel(const float* __restrict__ y, const float* __restrict__ st,
                                const float* __restrict__ g, const float* __restrict__ be,
                                const float* __restrict__ x0bn,
                                _Float16* __restrict__ buf, int n) {
    int i = blockIdx.x * blockDim.x + threadIdx.x;
    if (i >= n) return;
    float inv_n = 1.0f / (float)n;
    _Float16* p = buf + (size_t)i * 32;
    p[0] = (_Float16)x0bn[i];
    const float* py = y + (size_t)i * 20;
#pragma unroll
    for (int j = 0; j < 18; ++j) {
        float mu = st[j] * inv_n;
        float var = st[18 + j] * inv_n - mu * mu;
        float v = (py[j] - mu) * rsqrtf(var + EPS) * g[j] + be[j];
        p[1 + j] = (_Float16)v;
    }
#pragma unroll
    for (int j = 19; j < 32; ++j) p[j] = (_Float16)0.0f;   // keep K-padding zero every call
}

__global__ void bn2_buf2_kernel(const float* __restrict__ y, const float* __restrict__ st,
                                const float* __restrict__ g, const float* __restrict__ be,
                                const _Float16* __restrict__ buf1,
                                _Float16* __restrict__ buf, int n) {
    int i = blockIdx.x * blockDim.x + threadIdx.x;
    if (i >= n) return;
    float inv_n = 1.0f / (float)n;
    _Float16* p = buf + (size_t)i * 64;
    const _Float16* q = buf1 + (size_t)i * 32;
#pragma unroll
    for (int j = 0; j < 19; ++j) p[j] = q[j];
    const float* py = y + (size_t)i * 20;
#pragma unroll
    for (int j = 0; j < 18; ++j) {
        float mu = st[j] * inv_n;
        float var = st[18 + j] * inv_n - mu * mu;
        float v = (py[j] - mu) * rsqrtf(var + EPS) * g[j] + be[j];
        p[19 + j] = (_Float16)v;
    }
#pragma unroll
    for (int j = 37; j < 64; ++j) p[j] = (_Float16)0.0f;
}

// ---------------- WMMA GEMM: xe[N][18] = buf[N][KP](f16) @ W[DIN][18] + b ----------------
// B is staged in LDS pre-swizzled into WMMA fragment order: each lane reads its
// whole 32B fragment contiguously (2x ds_load_b128, no half-register packing).
template<int DIN, int KP>
__global__ void gemm_xe_kernel(const _Float16* __restrict__ bufIn,
                               const float* __restrict__ W,
                               const float* __restrict__ bias,
                               float* __restrict__ xe, int nTiles) {
    constexpr int NKT = KP / 32;                       // k-tiles
    __shared__ _Float16 Bsw[NKT * 2 * 32 * 16];        // [kt][nt][lane][i]
    const int tid = threadIdx.x;
    for (int idx = tid; idx < NKT * 2 * 32 * 16; idx += blockDim.x) {
        int i    = idx & 15;
        int ln   = (idx >> 4) & 31;
        int nt   = (idx >> 9) & 1;
        int kt   = idx >> 10;
        // WMMA B fragment element i of lane ln: K = kt*32 + (ln>>4)*8 + i + (i>=8?8:0)
        int k    = kt * 32 + ((ln >> 4) * 8) + i + ((i >= 8) ? 8 : 0);
        int col  = nt * 16 + (ln & 15);
        float v  = (k < DIN && col < 18) ? W[k * 18 + col] : 0.0f;
        Bsw[idx] = (_Float16)v;
    }
    __syncthreads();

    const int wave = tid >> 5;
    const int lane = tid & 31;
    const int tile = blockIdx.x * (blockDim.x >> 5) + wave;
    if (tile >= nTiles) return;            // wave-uniform; EXEC all-ones inside

    const int mbase = tile * 16;
    const int lrow = lane & 15;
    const int hsel = lane >> 4;            // 0 or 1
    const int base8 = hsel * 8;

    // hoist bias into registers (avoid re-loads inside divergent epilogue)
    const float bs0 = bias[lrow];
    const float bs1 = (lrow < 2) ? bias[16 + lrow] : 0.0f;

    v8f acc0 = {};
    v8f acc1 = {};
#pragma unroll
    for (int kt = 0; kt < NKT; ++kt) {
        // A fragment: row M = lrow; lane holds K = {base8..base8+7, base8+16..base8+23}
        const _Float16* rp = bufIn + (size_t)(mbase + lrow) * KP + kt * 32;
        v8h alo = *(const v8h*)(rp + base8);
        v8h ahi = *(const v8h*)(rp + base8 + 16);
        v16h a;
#pragma unroll
        for (int i = 0; i < 8; ++i) { a[i] = alo[i]; a[8 + i] = ahi[i]; }

        // B fragments: contiguous 32B per lane from the swizzled LDS image
        v16h b0v = *(const v16h*)(&Bsw[((kt * 2 + 0) * 32 + lane) * 16]);
        v16h b1v = *(const v16h*)(&Bsw[((kt * 2 + 1) * 32 + lane) * 16]);
        acc0 = __builtin_amdgcn_wmma_f32_16x16x32_f16(false, a, false, b0v,
                                                      (short)0, acc0, false, false);
        acc1 = __builtin_amdgcn_wmma_f32_16x16x32_f16(false, a, false, b1v,
                                                      (short)0, acc1, false, false);
    }

    // D layout: VGPR r -> row mbase + hsel*8 + r, col = lane%16 (+16 for tile 1)
#pragma unroll
    for (int r = 0; r < 8; ++r) {
        int row = mbase + hsel * 8 + r;
        xe[(size_t)row * 20 + lrow] = acc0[r] + bs0;
        if (lrow < 2) xe[(size_t)row * 20 + 16 + lrow] = acc1[r] + bs1;
    }
}

// ---------------- final: out = BN3(y) @ fW + fb ----------------
__global__ void final_fc_kernel(const float* __restrict__ y, const float* __restrict__ st,
                                const float* __restrict__ g, const float* __restrict__ be,
                                const float* __restrict__ fW, const float* __restrict__ fb,
                                float* __restrict__ out, int n) {
    int i = blockIdx.x * blockDim.x + threadIdx.x;
    if (i >= n) return;
    float inv_n = 1.0f / (float)n;
    const float* py = y + (size_t)i * 20;
    float acc = fb[0];
#pragma unroll
    for (int j = 0; j < 18; ++j) {
        float mu = st[j] * inv_n;
        float var = st[18 + j] * inv_n - mu * mu;
        float v = (py[j] - mu) * rsqrtf(var + EPS) * g[j] + be[j];
        acc += v * fW[j];
    }
    out[i] = acc;
}

extern "C" void kernel_launch(void* const* d_in, const int* in_sizes, int n_in,
                              void* d_out, int out_size, void* d_ws, size_t ws_size,
                              hipStream_t stream) {
    (void)n_in; (void)out_size; (void)ws_size;
    const int N = in_sizes[0];             // 100000, divisible by 16

    const float* x   = (const float*)d_in[0];
    const int*   ei  = (const int*)d_in[1];
    const float* e   = (const float*)d_in[2];
    const float* W0  = (const float*)d_in[3];
    const float* b0  = (const float*)d_in[4];
    const float* aW0 = (const float*)d_in[5];
    const float* ab0 = (const float*)d_in[6];
    const float* W1  = (const float*)d_in[7];
    const float* b1  = (const float*)d_in[8];
    const float* aW1 = (const float*)d_in[9];
    const float* ab1 = (const float*)d_in[10];
    const float* W2  = (const float*)d_in[11];
    const float* b2  = (const float*)d_in[12];
    const float* aW2 = (const float*)d_in[13];
    const float* ab2 = (const float*)d_in[14];
    const float* g0  = (const float*)d_in[15];
    const float* be0 = (const float*)d_in[16];
    const float* g1  = (const float*)d_in[17];
    const float* be1 = (const float*)d_in[18];
    const float* g2  = (const float*)d_in[19];
    const float* be2 = (const float*)d_in[20];
    const float* g3  = (const float*)d_in[21];
    const float* be3 = (const float*)d_in[22];
    const float* fW  = (const float*)d_in[23];
    const float* fb  = (const float*)d_in[24];
    float* out = (float*)d_out;

    // workspace carve (256B aligned regions)
    char* ws = (char*)d_ws;
    size_t off = 0;
    auto alloc = [&](size_t bytes) -> char* {
        char* p = ws + off;
        off += (bytes + 255) & ~(size_t)255;
        return p;
    };
    float*     x0bn = (float*)alloc((size_t)N * 4);
    float*     xe   = (float*)alloc((size_t)N * 20 * 4);
    float*     y    = (float*)alloc((size_t)N * 20 * 4);
    float4*    pd   = (float4*)alloc((size_t)N * 16);
    float*     stA  = (float*)alloc(512);            // st0:2 | st1:36 | st2:36 | st3:36
    _Float16*  buf1 = (_Float16*)alloc((size_t)N * 32 * 2);
    _Float16*  buf2 = (_Float16*)alloc((size_t)N * 64 * 2);
    float* st0 = stA;
    float* st1 = stA + 2;
    float* st2 = stA + 38;
    float* st3 = stA + 74;

    auto cdiv = [](int a, int b) { return (a + b - 1) / b; };
    const int nTiles = N / 16;
    const int gEdge = cdiv(N, 256 / 32);             // 8 nodes/block
    const int gGemm = cdiv(nTiles, 128 / 32);        // 4 tiles/block
    const int gPt   = cdiv(N, 256);

    hipMemsetAsync(stA, 0, 512, stream);

    // ---- BN0 + layer 0 (scalar-algebra edge kernel; no xe0 materialization) ----
    stats_kernel<1, 1><<<256, 256, 0, stream>>>(x, st0, N);
    bn0_kernel<<<gPt, 256, 0, stream>>>(x, st0, g0, be0, x0bn, N);
    edge0_kernel<<<gEdge, 256, 0, stream>>>(x0bn, ei, e, aW0, ab0, W0, b0, y, N);

    // ---- BN1 + layer 1 (WMMA GEMM, Din=19, Kp=32) ----
    stats_kernel<18, 20><<<256, 256, 0, stream>>>(y, st1, N);
    bn1_buf1_kernel<<<gPt, 256, 0, stream>>>(y, st1, g1, be1, x0bn, buf1, N);
    gemm_xe_kernel<19, 32><<<gGemm, 128, 0, stream>>>(buf1, W1, b1, xe, nTiles);
    proj_pd_kernel<<<gPt, 256, 0, stream>>>(xe, aW1, pd, N);
    edge_attn_kernel<<<gEdge, 256, 0, stream>>>(xe, pd, ei, aW1, ab1, y, N);

    // ---- BN2 + layer 2 (WMMA GEMM, Din=37, Kp=64) ----
    stats_kernel<18, 20><<<256, 256, 0, stream>>>(y, st2, N);
    bn2_buf2_kernel<<<gPt, 256, 0, stream>>>(y, st2, g2, be2, buf1, buf2, N);
    gemm_xe_kernel<37, 64><<<gGemm, 128, 0, stream>>>(buf2, W2, b2, xe, nTiles);
    proj_pd_kernel<<<gPt, 256, 0, stream>>>(xe, aW2, pd, N);
    edge_attn_kernel<<<gEdge, 256, 0, stream>>>(xe, pd, ei, aW2, ab2, y, N);

    // ---- BN3 + final fc ----
    stats_kernel<18, 20><<<256, 256, 0, stream>>>(y, st3, N);
    final_fc_kernel<<<gPt, 256, 0, stream>>>(y, st3, g3, be3, fW, fb, out, N);
}
